// HGRNet_10754598109735
// MI455X (gfx1250) — matrix operand order repaired and checked by reading, not compiled, not run on hardware
//
#include <hip/hip_runtime.h>
#include <hip/hip_bf16.h>

// ---------------------------------------------------------------------------
// HGNN (2x hyperconv + mean + FC) for MI455X / gfx1250.
// GEMMs run on v_wmma_f32_16x16x32_bf16 (wave32, f32 accum).
// Gather/scatter stages are L2-resident (whole working set ~125MB < 192MB L2).
// ---------------------------------------------------------------------------

typedef __attribute__((ext_vector_type(16))) __bf16 v16bf;
typedef __attribute__((ext_vector_type(8)))  __bf16 v8bf;
typedef __attribute__((ext_vector_type(8)))  float  v8f;

#define KNN 10
#define NEG_SLOPE 0.01f

// First K-1=9 entries of np.random.RandomState(0).permutation(19).
// NOTE: compile-only environment -- value set is a deterministic placeholder;
// swap in the exact host permutation if runtime validation becomes available.
__device__ __constant__ int c_perm[KNN - 1] = {13, 2, 17, 5, 8, 11, 0, 15, 7};

// ---------------------------------------------------------------------------
// Hypergraph construction: global node index per (edge, slot) + dv counts.
// gidx entries are flattened b*N + node so downstream kernels ignore batch.
// ---------------------------------------------------------------------------
__global__ void build_idx_kernel(const int* __restrict__ nn, int* __restrict__ gidx,
                                 int* __restrict__ dvcnt, int B, int N, int NNW) {
    int t = blockIdx.x * blockDim.x + threadIdx.x;   // over B*N edges
    if (t >= B * N) return;
    int b = t / N;
    const int* row = nn + (size_t)t * NNW;
    int base = t * KNN;
    gidx[base] = t;                                   // self member
    atomicAdd(&dvcnt[t], 1);
#pragma unroll
    for (int j = 0; j < KNN - 1; ++j) {
        int g = b * N + row[c_perm[j]];
        gidx[base + 1 + j] = g;
        atomicAdd(&dvcnt[g], 1);
    }
}

__global__ void dvinv_kernel(const int* __restrict__ dvcnt, float* __restrict__ dvinv, int BN) {
    int t = blockIdx.x * blockDim.x + threadIdx.x;
    if (t < BN) dvinv[t] = 1.0f / fmaxf((float)dvcnt[t], 1.0f);
}

// ---------------------------------------------------------------------------
// f32 -> bf16 conversions (X, and Theta transposed to K-contiguous columns).
// ---------------------------------------------------------------------------
__global__ void cvt_x_kernel(const float* __restrict__ x, __bf16* __restrict__ xb, int total) {
    int t = blockIdx.x * blockDim.x + threadIdx.x;
    if (t < total) xb[t] = (__bf16)x[t];
}

__global__ void cvt_theta_kernel(const float* __restrict__ th, __bf16* __restrict__ wt,
                                 int Kdim, int H) {
    int t = blockIdx.x * blockDim.x + threadIdx.x;   // t = k*H + h
    if (t >= Kdim * H) return;
    int k = t / H, h = t - k * H;
    wt[(size_t)h * Kdim + k] = (__bf16)th[t];        // Wt: (H, K) row-major
}

// ---------------------------------------------------------------------------
// GEMM: C(BN x H) = A(BN x K, bf16) * Wt^T (Wt stored H x K, bf16), f32 accum.
// One wave computes a 16-row x 64-col strip: A-fragment reused across 4 WMMAs.
// Fragment loads follow the CDNA5 16-bit A / B VGPR layouts:
//   A (16x32): lane L<16 holds row M=L, K = {0..7, 16..23}; L>=16 -> +8.
//   B (32x16): lane L<16 holds col N=L, K = {0..7, 16..23}; L>=16 -> +8.
// All fragment loads are contiguous 16B (global_load_b128).
// ---------------------------------------------------------------------------
__global__ void __launch_bounds__(128)
gemm_bf16_wmma_kernel(const __bf16* __restrict__ A, const __bf16* __restrict__ Wt,
                      float* __restrict__ C, int Kdim, int H) {
    const int lane  = threadIdx.x;            // 0..31 (wave32)
    const int wave  = threadIdx.y;            // 64-column group
    const int row16 = blockIdx.x * 16;
    const int col0  = wave * 64;
    const int lo    = lane & 15;
    const int hiA   = (lane >> 4) * 8;        // A K-half select
    const int hiB   = (lane >> 4) * 8;        // B K-half select

    const __bf16* Arow = A + (size_t)(row16 + lo) * Kdim;

    v8f acc[4];
#pragma unroll
    for (int s = 0; s < 4; ++s) acc[s] = (v8f){0.f, 0.f, 0.f, 0.f, 0.f, 0.f, 0.f, 0.f};

    for (int k0 = 0; k0 < Kdim; k0 += 32) {
        union { v16bf v; v8bf h[2]; } afrag;
        afrag.h[0] = *(const v8bf*)(Arow + k0 + hiA);        // K = k0+hiA .. +7
        afrag.h[1] = *(const v8bf*)(Arow + k0 + 16 + hiA);   // K = k0+16+hiA .. +7
#pragma unroll
        for (int s = 0; s < 4; ++s) {
            const __bf16* Wcol = Wt + (size_t)(col0 + s * 16 + lo) * Kdim + k0 + hiB;
            union { v16bf v; v8bf h[2]; } bfrag;
            bfrag.h[0] = *(const v8bf*)(Wcol);
            bfrag.h[1] = *(const v8bf*)(Wcol + 16);
            acc[s] = __builtin_amdgcn_wmma_f32_16x16x32_bf16(
                /*neg_a=*/false, afrag.v, /*neg_b=*/false, bfrag.v,
                /*c_mod=*/(short)0, acc[s], /*reuse_a=*/false, /*reuse_b=*/false);
        }
    }

    // C/D layout: VGPR j -> row M = j + 8*(lane>=16), col N = lane&15.
    const int hi = lane >> 4;
#pragma unroll
    for (int s = 0; s < 4; ++s)
#pragma unroll
        for (int j = 0; j < 8; ++j)
            C[(size_t)(row16 + j + 8 * hi) * H + col0 + s * 16 + lo] = acc[s][j];
}

// ---------------------------------------------------------------------------
// EF[e,:] = (1/K) * sum over the edge's 10 member rows of Y. One block/edge,
// one lane/channel -> each member row is a coalesced 1KB (H1) / 512B (H2) read.
// ---------------------------------------------------------------------------
__global__ void edge_feat_kernel(const float* __restrict__ Y, const int* __restrict__ gidx,
                                 float* __restrict__ EF, int H) {
    int e = blockIdx.x, c = threadIdx.x;
    const int* idx = gidx + (size_t)e * KNN;
    float s = 0.f;
#pragma unroll
    for (int j = 0; j < KNN; ++j) s += Y[(size_t)idx[j] * H + c];
    EF[(size_t)e * H + c] = s * (1.0f / KNN);
}

// OUT[v,:] += EF[e,:] / dv[v] for every incidence. f32 atomics resolved in L2.
__global__ void node_agg_kernel(const float* __restrict__ EF, const int* __restrict__ gidx,
                                const float* __restrict__ dvinv, float* __restrict__ OUT, int H) {
    int e = blockIdx.x, c = threadIdx.x;
    float ef = EF[(size_t)e * H + c];
    const int* idx = gidx + (size_t)e * KNN;
#pragma unroll
    for (int j = 0; j < KNN; ++j) {
        int v = idx[j];
        atomicAdd(&OUT[(size_t)v * H + c], ef * dvinv[v]);
    }
}

// bias + leaky relu, emitting bf16 (next layer's A matrix) or f32 (final feats)
__global__ void act_bf16_kernel(const float* __restrict__ in, const float* __restrict__ bias,
                                __bf16* __restrict__ out, int H, int total) {
    int t = blockIdx.x * blockDim.x + threadIdx.x;
    if (t >= total) return;
    float o = in[t] + bias[t % H];
    o = (o > 0.f) ? o : NEG_SLOPE * o;
    out[t] = (__bf16)o;
}

__global__ void act_f32_kernel(const float* __restrict__ in, const float* __restrict__ bias,
                               float* __restrict__ out, int H, int total) {
    int t = blockIdx.x * blockDim.x + threadIdx.x;
    if (t >= total) return;
    float o = in[t] + bias[t % H];
    out[t] = (o > 0.f) ? o : NEG_SLOPE * o;
}

// mean over N nodes per (batch, channel); one block per (b,c), LDS tree reduce.
__global__ void mean_nodes_kernel(const float* __restrict__ F, float* __restrict__ meanbuf,
                                  int N, int H) {
    __shared__ float red[256];
    int bc = blockIdx.x;
    int b = bc / H, c = bc - b * H;
    const float* base = F + (size_t)b * N * H + c;
    float s = 0.f;
    for (int n = threadIdx.x; n < N; n += blockDim.x) s += base[(size_t)n * H];
    red[threadIdx.x] = s;
    __syncthreads();
    for (int off = 128; off > 0; off >>= 1) {
        if ((int)threadIdx.x < off) red[threadIdx.x] += red[threadIdx.x + off];
        __syncthreads();
    }
    if (threadIdx.x == 0) meanbuf[bc] = red[0] * (1.0f / N);
}

// final FC: (B,H2) @ (H2,T) + bias. B*T = 256 = exactly one block, f32 math.
__global__ void fc_kernel(const float* __restrict__ meanbuf, const float* __restrict__ w,
                          const float* __restrict__ bias, float* __restrict__ out,
                          int H, int T) {
    int t = threadIdx.x;           // 0..255
    int b = t / T, o = t - b * T;
    float s = bias[o];
    for (int c = 0; c < H; ++c) s += meanbuf[b * H + c] * w[c * T + o];
    out[b * T + o] = s;
}

// ---------------------------------------------------------------------------
extern "C" void kernel_launch(void* const* d_in, const int* in_sizes, int n_in,
                              void* d_out, int out_size, void* d_ws, size_t ws_size,
                              hipStream_t stream) {
    (void)in_sizes; (void)n_in; (void)out_size; (void)ws_size;
    const int B = 4, N = 10000, Cin = 512, H1 = 256, H2 = 128, T = 64;
    const int BN = B * N;

    const float* x      = (const float*)d_in[0];
    const int*   nn_idx = (const int*)  d_in[1];
    const float* theta0 = (const float*)d_in[2];
    const float* b0     = (const float*)d_in[3];
    const float* theta1 = (const float*)d_in[4];
    const float* b1     = (const float*)d_in[5];
    const float* fc_w   = (const float*)d_in[6];
    const float* fc_b   = (const float*)d_in[7];
    float* out = (float*)d_out;

    // ---- workspace carving (256B-aligned slabs; ~125MB total, L2-resident) --
    char* ws = (char*)d_ws;
    size_t off = 0;
    auto carve = [&](size_t bytes) -> char* {
        char* p = ws + off;
        off += (bytes + 255) & ~(size_t)255;
        return p;
    };
    __bf16* Xb      = (__bf16*)carve((size_t)BN * Cin * sizeof(__bf16)); // layer1 A; reused as layer2 A (BN x H1 bf16)
    float*  bufY    = (float*) carve((size_t)BN * H1 * sizeof(float));   // GEMM out; reused as OUT accumulator
    float*  bufEF   = (float*) carve((size_t)BN * H1 * sizeof(float));   // edge feats; reused as final f32 feats
    __bf16* Wt      = (__bf16*)carve((size_t)Cin * H1 * sizeof(__bf16)); // transposed theta (bf16), reused per layer
    int*    gidx    = (int*)   carve((size_t)BN * KNN * sizeof(int));
    int*    dvcnt   = (int*)   carve((size_t)BN * sizeof(int));
    float*  dvinv   = (float*) carve((size_t)BN * sizeof(float));
    float*  meanbuf = (float*) carve((size_t)B * H2 * sizeof(float));

    const int TPB = 256;

    // ---- hypergraph structure ----------------------------------------------
    hipMemsetAsync(dvcnt, 0, (size_t)BN * sizeof(int), stream);
    build_idx_kernel<<<(BN + TPB - 1) / TPB, TPB, 0, stream>>>(nn_idx, gidx, dvcnt, B, N, 2 * KNN - 1);
    dvinv_kernel<<<(BN + TPB - 1) / TPB, TPB, 0, stream>>>(dvcnt, dvinv, BN);

    // ---- layer 1 -----------------------------------------------------------
    cvt_x_kernel<<<((size_t)BN * Cin) / TPB, TPB, 0, stream>>>(x, Xb, BN * Cin);
    cvt_theta_kernel<<<(Cin * H1 + TPB - 1) / TPB, TPB, 0, stream>>>(theta0, Wt, Cin, H1);
    gemm_bf16_wmma_kernel<<<dim3(BN / 16), dim3(32, H1 / 64), 0, stream>>>(Xb, Wt, bufY, Cin, H1);
    edge_feat_kernel<<<BN, H1, 0, stream>>>(bufY, gidx, bufEF, H1);
    hipMemsetAsync(bufY, 0, (size_t)BN * H1 * sizeof(float), stream);
    node_agg_kernel<<<BN, H1, 0, stream>>>(bufEF, gidx, dvinv, bufY, H1);
    act_bf16_kernel<<<((size_t)BN * H1) / TPB, TPB, 0, stream>>>(bufY, b0, Xb, H1, BN * H1);

    // ---- layer 2 -----------------------------------------------------------
    cvt_theta_kernel<<<(H1 * H2 + TPB - 1) / TPB, TPB, 0, stream>>>(theta1, Wt, H1, H2);
    gemm_bf16_wmma_kernel<<<dim3(BN / 16), dim3(32, H2 / 64), 0, stream>>>(Xb, Wt, bufY, H1, H2);
    edge_feat_kernel<<<BN, H2, 0, stream>>>(bufY, gidx, bufEF, H2);
    hipMemsetAsync(bufY, 0, (size_t)BN * H2 * sizeof(float), stream);
    node_agg_kernel<<<BN, H2, 0, stream>>>(bufEF, gidx, dvinv, bufY, H2);
    act_f32_kernel<<<((size_t)BN * H2) / TPB, TPB, 0, stream>>>(bufY, b1, bufEF, H2, BN * H2);

    // ---- pooling + FC ------------------------------------------------------
    mean_nodes_kernel<<<B * H2, 256, 0, stream>>>(bufEF, meanbuf, N, H2);
    fc_kernel<<<1, B * T, 0, stream>>>(meanbuf, fc_w, fc_b, out, H2, T);
}
